// PairNN_5403068858494
// MI455X (gfx1250) — matrix operand order repaired and checked by reading, not compiled, not run on hardware
//
#include <hip/hip_runtime.h>
#include <hip/hip_bf16.h>

typedef __attribute__((ext_vector_type(16))) _Float16 v16h;
typedef __attribute__((ext_vector_type(8)))  _Float16 v8h;
typedef __attribute__((ext_vector_type(8)))  float    v8f;

// Problem constants (fixed by the reference)
#define LQ     128
#define KW     8
#define NPAIR  2104          // windowed pairs for L=128, K=8
#define BATCH  16
#define MTOT   (BATCH * NPAIR)   // 33664, divisible by 128
#define GDIM   512
#define FDIM   1088          // 2*512 + 64
#define NTILES 17            // 1088 / 64

// ---------------------------------------------------------------------------
// rowstart(i): number of pairs in rows < i (closed form)
// ---------------------------------------------------------------------------
__device__ __forceinline__ int rowstart(int i) {
    if (i <= KW) return i * (KW + 1) + (i * (i - 1)) / 2;
    int rs = KW * (KW + 1) + (KW * (KW - 1)) / 2;          // rowstart(K) = 100
    if (i <= LQ - KW) return rs + (i - KW) * (2 * KW + 1);
    rs += (LQ - 2 * KW) * (2 * KW + 1);                    // rowstart(120) = 2004
    int a = i - (LQ - KW);
    return rs + a * 2 * KW - (a * (a - 1)) / 2;
}

// ---------------------------------------------------------------------------
// Kernel 1: pair table + integer-position output
// ---------------------------------------------------------------------------
__global__ void pairs_kernel(int4* __restrict__ pairs, float* __restrict__ out_pos) {
    int i = blockIdx.x;           // 0..127
    int t = threadIdx.x;          // 0..31
    int lo = max(i - KW, 0), hi = min(i + KW, LQ - 1);
    int cnt = hi - lo + 1;
    if (t < cnt) {
        int j = lo + t;
        int p = rowstart(i) + t;
        pairs[p] = make_int4(i, j, j - i + KW, 0);   // store rel bucket (0..16)
        out_pos[2 * p]     = (float)(i + 1);
        out_pos[2 * p + 1] = (float)(j + 1);
    }
}

// ---------------------------------------------------------------------------
// Kernel 2: bucketed Gaussian smoothing: 17 x 17 x 64 (replaces n^2 x 64)
// smoothed[r][e] = sum_{r'} (L - |r'-K|) * exp(-(r-r')^2) * pos_emb[r'][e]
// ---------------------------------------------------------------------------
__global__ void smooth_kernel(const float* __restrict__ pos_emb,
                              float* __restrict__ smoothed) {
    int r = blockIdx.x;           // 0..16
    int e = threadIdx.x;          // 0..63
    float s = 0.f;
#pragma unroll
    for (int rp = 0; rp < 17; ++rp) {
        float cnt = (float)(LQ - abs(rp - KW));
        float d = (float)(r - rp);
        s += cnt * expf(-d * d) * pos_emb[rp * 64 + e];
    }
    smoothed[r * 64 + e] = s;
}

// ---------------------------------------------------------------------------
// Kernel 3: W1 -> f16 (row-major [g][f], exactly the B-operand layout we need)
// ---------------------------------------------------------------------------
__global__ void w1_to_half(const float* __restrict__ W1,
                           _Float16* __restrict__ W1h, int n) {
    int i = blockIdx.x * blockDim.x + threadIdx.x;
    if (i < n) W1h[i] = (_Float16)W1[i];
}

// ---------------------------------------------------------------------------
// Kernel 4: fused gather + f32->f16 + WMMA GEMM + bias + ReLU + W2 partial dot
// Block: 256 threads (8 wave32), tile 128(M) x 64(N), BK=32.
// K-loop statically split into 3 phases (emo / cause / smoothed) so the A
// source pointer is loop-invariant per phase -> no per-iteration branching.
// Waves 4x2; each wave computes a 32x32 region = 2x2 v_wmma_f32_16x16x32_f16.
// Writes partial[m*17 + bn] (race-free; deterministic final reduce).
// ---------------------------------------------------------------------------
__global__ __launch_bounds__(256)
void pairnn_gemm(const float* __restrict__ h_emo,
                 const float* __restrict__ h_cause,
                 const float* __restrict__ smoothed,
                 const _Float16* __restrict__ W1h,
                 const float* __restrict__ b1,
                 const float* __restrict__ W2,
                 const int4* __restrict__ pairs,
                 float* __restrict__ partial) {
    __shared__ _Float16 sA[128 * 32];   // 8 KB  A tile (rows m, 32 k-halves)
    __shared__ _Float16 sB[64 * 32];    // 4 KB  B tile (rows g, 32 k-halves)
    __shared__ float    sH[128 * 64];   // 32 KB h*w2 stage
    __shared__ float    sRed[256];

    const int t    = threadIdx.x;
    const int lane = t & 31;
    const int wave = t >> 5;
    const int wm   = wave & 3;          // 0..3 -> 32-row band
    const int wn   = wave >> 2;         // 0..1 -> 32-col band
    const int bm   = blockIdx.x;        // 0..262
    const int bn   = blockIdx.y;        // 0..16

    // ---- A-stage assignment: 2 threads per row, 16 halves each ----
    const int rowA = t >> 1;            // 0..127
    const int chA  = t & 1;             // 0..1
    const int m    = bm * 128 + rowA;
    const int bb   = m / NPAIR;
    const int p    = m - bb * NPAIR;
    const int4 pr  = pairs[p];
    // per-phase loop-invariant base pointers (chA offset folded in)
    const float* aEmo = h_emo   + (size_t)(bb * LQ + pr.x) * GDIM + chA * 16;
    const float* aCau = h_cause + (size_t)(bb * LQ + pr.y) * GDIM + chA * 16;
    const float* aSm  = smoothed + pr.z * 64 + chA * 16;

    // ---- B-stage assignment: 4 threads per row, 8 halves each ----
    const int rowB = t >> 2;            // 0..63
    const int chB  = t & 3;             // 0..3
    const _Float16* w1Row = W1h + (size_t)(bn * 64 + rowB) * FDIM + chB * 8;

    // LDS addresses (loop-invariant)
    _Float16* sAst = &sA[rowA * 32 + chA * 16];
    _Float16* sBst = &sB[rowB * 32 + chB * 8];
    const int lm = lane & 15;
    const int ka = (lane < 16) ? 0 : 8;   // A half-chunk base (ISA A layout)
    const int kb = (lane < 16) ? 0 : 8;   // B half-chunk base (ISA B layout)
    const _Float16* fragA0 = &sA[(wm * 32 + lm) * 32 + ka];
    const _Float16* fragA1 = fragA0 + 16 * 32;
    const _Float16* fragB0 = &sB[(wn * 32 + lm) * 32 + kb];
    const _Float16* fragB1 = fragB0 + 16 * 32;

    const v8f vzero = {0.f, 0.f, 0.f, 0.f, 0.f, 0.f, 0.f, 0.f};
    v8f acc[2][2];
#pragma unroll
    for (int i = 0; i < 2; ++i)
#pragma unroll
        for (int j = 0; j < 2; ++j) acc[i][j] = vzero;

    // one GEMM mainloop step: fixed A source pointer, no control flow
    auto gemm_step = [&](const float* __restrict__ asrc,
                         const _Float16* __restrict__ bsrcp) {
        const float4 x0 = ((const float4*)asrc)[0];
        const float4 x1 = ((const float4*)asrc)[1];
        const float4 x2 = ((const float4*)asrc)[2];
        const float4 x3 = ((const float4*)asrc)[3];
        const uint4 bsrc = *(const uint4*)bsrcp;

        v8h alo, ahi;
        alo[0] = (_Float16)x0.x; alo[1] = (_Float16)x0.y;
        alo[2] = (_Float16)x0.z; alo[3] = (_Float16)x0.w;
        alo[4] = (_Float16)x1.x; alo[5] = (_Float16)x1.y;
        alo[6] = (_Float16)x1.z; alo[7] = (_Float16)x1.w;
        ahi[0] = (_Float16)x2.x; ahi[1] = (_Float16)x2.y;
        ahi[2] = (_Float16)x2.z; ahi[3] = (_Float16)x2.w;
        ahi[4] = (_Float16)x3.x; ahi[5] = (_Float16)x3.y;
        ahi[6] = (_Float16)x3.z; ahi[7] = (_Float16)x3.w;

        __syncthreads();   // previous iteration's fragment reads complete
        *(v8h*)(sAst)     = alo;
        *(v8h*)(sAst + 8) = ahi;
        *(uint4*)(sBst)   = bsrc;
        __syncthreads();

        v16h af[2], bf[2];
        {
            v8h* d = (v8h*)&af[0];
            d[0] = *(const v8h*)(fragA0);
            d[1] = *(const v8h*)(fragA0 + 16);
        }
        {
            v8h* d = (v8h*)&af[1];
            d[0] = *(const v8h*)(fragA1);
            d[1] = *(const v8h*)(fragA1 + 16);
        }
        {
            v8h* d = (v8h*)&bf[0];
            d[0] = *(const v8h*)(fragB0);
            d[1] = *(const v8h*)(fragB0 + 16);
        }
        {
            v8h* d = (v8h*)&bf[1];
            d[0] = *(const v8h*)(fragB1);
            d[1] = *(const v8h*)(fragB1 + 16);
        }
#pragma unroll
        for (int i = 0; i < 2; ++i)
#pragma unroll
            for (int j = 0; j < 2; ++j)
                acc[i][j] = __builtin_amdgcn_wmma_f32_16x16x32_f16(
                    false, af[i], false, bf[j], (short)0, acc[i][j],
                    false, false);
    };

    // Phase 1: emotion features, f in [0, 512) -> kt 0..15
    for (int kt = 0; kt < 16; ++kt)
        gemm_step(aEmo + kt * 32, w1Row + kt * 32);
    // Phase 2: cause features, f in [512, 1024) -> kt 16..31
    for (int kt = 0; kt < 16; ++kt)
        gemm_step(aCau + kt * 32, w1Row + (16 + kt) * 32);
    // Phase 3: smoothed rel-embedding, f in [1024, 1088) -> kt 32..33
#pragma unroll
    for (int kt = 0; kt < 2; ++kt)
        gemm_step(aSm + kt * 32, w1Row + (32 + kt) * 32);

    // -------- epilogue: +b1, ReLU, *W2, stage to LDS --------
    const int hi8 = (lane < 16) ? 0 : 8;
#pragma unroll
    for (int j = 0; j < 2; ++j) {
        const int colLocal = wn * 32 + j * 16 + lm;
        const int g = bn * 64 + colLocal;
        const float bias = b1[g];
        const float w2v  = W2[g];
#pragma unroll
        for (int i = 0; i < 2; ++i) {
            const int rbase = wm * 32 + i * 16 + hi8;
#pragma unroll
            for (int v = 0; v < 8; ++v) {
                float hv = acc[i][j][v] + bias;
                hv = fmaxf(hv, 0.0f);
                sH[(rbase + v) * 64 + colLocal] = hv * w2v;
            }
        }
    }
    __syncthreads();

    // row-reduce the 64-wide h*w2 stage
    {
        const int row = t >> 1, half = t & 1;
        const float* hr = &sH[row * 64 + half * 32];
        float s = 0.f;
#pragma unroll
        for (int c = 0; c < 32; ++c) s += hr[c];
        sRed[t] = s;
    }
    __syncthreads();
    if (t < 128)
        partial[(size_t)(bm * 128 + t) * NTILES + bn] = sRed[2 * t] + sRed[2 * t + 1];
}

// ---------------------------------------------------------------------------
// Kernel 5: deterministic final reduction over the 17 N-tiles, + b2
// ---------------------------------------------------------------------------
__global__ void reduce_kernel(const float* __restrict__ partial,
                              const float* __restrict__ b2,
                              float* __restrict__ pred) {
    int m = blockIdx.x * blockDim.x + threadIdx.x;
    if (m < MTOT) {
        float s = b2[0];
        const float* pp = partial + (size_t)m * NTILES;
#pragma unroll
        for (int tn = 0; tn < NTILES; ++tn) s += pp[tn];
        pred[m] = s;
    }
}

// ---------------------------------------------------------------------------
// Workspace layout (bytes):
//   [0)        int4 pairs[2104]            (33,664 B)
//   [36864)    float smoothed[17*64]       ( 4,352 B)
//   [49152)    _Float16 W1h[1088*1088]     (2,367,488 B)
//   [2424832)  float partial[33664*17]     (2,289,152 B)  -> total < 5 MB
// ---------------------------------------------------------------------------
extern "C" void kernel_launch(void* const* d_in, const int* in_sizes, int n_in,
                              void* d_out, int out_size, void* d_ws, size_t ws_size,
                              hipStream_t stream) {
    (void)in_sizes; (void)n_in; (void)out_size; (void)ws_size;
    const float* h_emo   = (const float*)d_in[0];
    const float* h_cause = (const float*)d_in[1];
    const float* pos_emb = (const float*)d_in[2];
    const float* W1      = (const float*)d_in[3];
    const float* b1      = (const float*)d_in[4];
    const float* W2      = (const float*)d_in[5];
    const float* b2      = (const float*)d_in[6];
    float* out = (float*)d_out;

    char* ws = (char*)d_ws;
    int4*     pairs    = (int4*)(ws + 0);
    float*    smoothed = (float*)(ws + 36864);
    _Float16* W1h      = (_Float16*)(ws + 49152);
    float*    partial  = (float*)(ws + 2424832);

    pairs_kernel<<<128, 32, 0, stream>>>(pairs, out + MTOT);
    smooth_kernel<<<17, 64, 0, stream>>>(pos_emb, smoothed);
    w1_to_half<<<(FDIM * FDIM + 255) / 256, 256, 0, stream>>>(W1, W1h, FDIM * FDIM);
    pairnn_gemm<<<dim3(MTOT / 128, NTILES), 256, 0, stream>>>(
        h_emo, h_cause, smoothed, W1h, b1, W2, pairs, partial);
    reduce_kernel<<<(MTOT + 255) / 256, 256, 0, stream>>>(partial, b2, out);
}
